// MoCoSuperpixelCluster_34119220199588
// MI455X (gfx1250) — matrix operand order repaired and checked by reading, not compiled, not run on hardware
//
#include <hip/hip_runtime.h>
#include <math.h>

// ---------------- problem constants ----------------
#define BATCH   256
#define HIDDEN  2048
#define DIM     128
#define QUEUE_N 65536
#define KCLUST  50
#define N_HARD  1024
#define N_FALSE 256

// ---------------- output element offsets (float32/int32 slots) ----------------
#define O_Q      0
#define O_K1     (BATCH*DIM)
#define O_K2     (2*BATCH*DIM)
#define O_FEMB   (3*BATCH*DIM)
#define O_HEMB   (O_FEMB + BATCH*N_FALSE*DIM)
#define O_FIDX   (O_HEMB + BATCH*N_HARD*DIM)
#define O_HIDX   (O_FIDX + BATCH*N_FALSE)
#define O_QIDS   (O_HIDX + BATCH*N_HARD)
#define O_K1IDS  (O_QIDS + BATCH)
#define O_K2IDS  (O_K1IDS + BATCH)
#define O_QUEUE  (O_K2IDS + BATCH)

// ---------------- workspace byte offsets ----------------
#define WS_XQ     ((size_t)0)
#define WS_XK1    ((size_t)1 << 20)
#define WS_XK2    ((size_t)2 << 20)
#define WS_W1Q    ((size_t)3 << 20)
#define WS_W1K    ((size_t)11 << 20)
#define WS_W2Q    ((size_t)19 << 20)
#define WS_W2K    (((size_t)19 << 20) + ((size_t)1 << 19))
#define WS_C1     ((size_t)20 << 20)
#define WS_H      ((size_t)22 << 20)
#define WS_C2     ((size_t)23 << 20)
#define WS_QBF    (((size_t)23 << 20) + ((size_t)1 << 18))
#define WS_QUEBF  ((size_t)24 << 20)
#define WS_SIMS   ((size_t)40 << 20)

typedef __bf16 bf16_t;
typedef __attribute__((ext_vector_type(16))) bf16_t bf16x16;
typedef __attribute__((ext_vector_type(8)))  bf16_t bf16x8;
typedef __attribute__((ext_vector_type(8)))  float  f32x8;

static __device__ __forceinline__ bf16_t f2bf(float f) {
    unsigned u = __float_as_uint(f);
    unsigned short h = (unsigned short)((u + 0x7FFFu + ((u >> 16) & 1u)) >> 16);
    bf16_t r; __builtin_memcpy(&r, &h, 2); return r;
}

// ---------------- f32 -> bf16 convert ----------------
__global__ void cvt_bf16_kernel(const float* __restrict__ in, bf16_t* __restrict__ out, int n) {
    for (int i = blockIdx.x * blockDim.x + threadIdx.x; i < n; i += gridDim.x * blockDim.x)
        out[i] = f2bf(in[i]);
}

// in [K,N] f32 row-major -> out [N,K] bf16 row-major (pre-transposed B for WMMA)
__global__ void cvt_bf16_T_kernel(const float* __restrict__ in, bf16_t* __restrict__ out,
                                  int K, int N) {
    int total = K * N;
    for (int i = blockIdx.x * blockDim.x + threadIdx.x; i < total; i += gridDim.x * blockDim.x) {
        int n = i / K, k = i - n * K;
        out[i] = f2bf(in[(size_t)k * N + n]);
    }
}

// ---------------- WMMA GEMM: C[M,N] = A[M,K] * BT[N,K]^T  (bf16 in, f32 out) ----------------
// 128 threads = 4 waves. Block tile: 16 (M) x (64*NT) (N). Wave w owns NT 16x16 N-tiles,
// reusing one A fragment per K-step. The block's 16xK A slab (contiguous in global memory)
// is staged into LDS with GLOBAL_LOAD_ASYNC_TO_LDS_B128 (ASYNCcnt) before the K loop.
template <int K, int NT>
__global__ void wmma_gemm_bf16_kernel(const bf16_t* __restrict__ A,
                                      const bf16_t* __restrict__ BT,
                                      float* __restrict__ C, int N) {
    __shared__ __align__(16) bf16_t smemA[16 * K];

    const int lane   = threadIdx.x & 31;
    const int wave   = threadIdx.x >> 5;
    const int m_tile = blockIdx.y;
    const int n0tile = blockIdx.x * (4 * NT) + wave * NT;
    const int l16    = lane & 15;
    const int hi     = lane >> 4;          // 0: lanes 0-15, 1: lanes 16-31

    // ---- async-stage the contiguous 16xK A slab into LDS ----
    {
        const unsigned lds_base = (unsigned)(unsigned long long)(void*)smemA;
        const char* gbase = (const char*)(A + (size_t)m_tile * 16 * K);
        for (int c = threadIdx.x; c < 2 * K; c += 128) {   // 2*K chunks of 16 bytes
            unsigned dst = lds_base + c * 16;
            const char* src = gbase + c * 16;
            asm volatile("global_load_async_to_lds_b128 %0, %1, off"
                         :: "v"(dst), "v"(src) : "memory");
        }
        asm volatile("s_wait_asynccnt 0x0" ::: "memory");
        __syncthreads();
    }

    // B pointers: BT row n is B column n (contiguous); lanes 0-15 hold K 0-15, lanes 16-31 hold 16-31
    const bf16_t* pb[NT];
#pragma unroll
    for (int j = 0; j < NT; ++j)
        pb[j] = BT + (size_t)((n0tile + j) * 16 + l16) * K + hi * 16;

    // A fragment read from LDS, contiguous K order matching B:
    // lanes 0-15 hold K 0-15; lanes 16-31 hold K 16-31
    const bf16_t* sA = smemA + (size_t)l16 * K + hi * 16;

    f32x8 acc[NT];
#pragma unroll
    for (int j = 0; j < NT; ++j) acc[j] = (f32x8){};

#pragma unroll 4
    for (int k0 = 0; k0 < K; k0 += 32) {
        bf16x8 alo = *(const bf16x8*)(sA + k0);
        bf16x8 ahi = *(const bf16x8*)(sA + k0 + 8);
        bf16x16 a;
#pragma unroll
        for (int i = 0; i < 8; ++i) { a[i] = alo[i]; a[8 + i] = ahi[i]; }
#pragma unroll
        for (int j = 0; j < NT; ++j) {
            __builtin_prefetch(pb[j] + k0 + 64, 0, 0);
            bf16x16 b = *(const bf16x16*)(pb[j] + k0);
            acc[j] = __builtin_amdgcn_wmma_f32_16x16x32_bf16(false, a, false, b,
                                                             (short)0, acc[j], false, false);
        }
    }

    // C layout: VGPR i -> M = m_tile*16 + hi*8 + i, N = n_tile*16 + l16
    const int crow0 = m_tile * 16 + hi * 8;
#pragma unroll
    for (int j = 0; j < NT; ++j) {
        const int ccol = (n0tile + j) * 16 + l16;
#pragma unroll
        for (int i = 0; i < 8; ++i)
            C[(size_t)(crow0 + i) * N + ccol] = acc[j][i];
    }
}

// ---------------- epilogue: H = bf16(relu(C + bias)) ----------------
__global__ void bias_relu_bf16_kernel(const float* __restrict__ Cin, const float* __restrict__ bias,
                                      bf16_t* __restrict__ H, int total, int N) {
    for (int i = blockIdx.x * blockDim.x + threadIdx.x; i < total; i += gridDim.x * blockDim.x) {
        float v = Cin[i] + bias[i & (N - 1)];
        H[i] = f2bf(v > 0.f ? v : 0.f);
    }
}

// ---------------- epilogue: z = l2norm(C + bias) (one block per row, 128 threads) ----------------
__global__ void bias_l2norm_kernel(const float* __restrict__ Cin, const float* __restrict__ bias,
                                   float* __restrict__ zout, bf16_t* __restrict__ zbf) {
    const int row = blockIdx.x, t = threadIdx.x;
    __shared__ float red[DIM];
    float v = Cin[(size_t)row * DIM + t] + bias[t];
    red[t] = v * v;
    __syncthreads();
    for (int s = DIM / 2; s > 0; s >>= 1) {
        if (t < s) red[t] += red[t + s];
        __syncthreads();
    }
    float o = v / sqrtf(red[0]);
    zout[(size_t)row * DIM + t] = o;
    if (zbf) zbf[(size_t)row * DIM + t] = f2bf(o);
}

// ---------------- centroid argmin (one block per row, 128 threads) ----------------
__global__ void assign_centroids_kernel(const float* __restrict__ Z,
                                        const float* __restrict__ cent,
                                        int* __restrict__ ids) {
    const int row = blockIdx.x, t = threadIdx.x;
    __shared__ float zz[DIM];
    __shared__ float d2[64];
    zz[t] = Z[(size_t)row * DIM + t];
    __syncthreads();
    if (t < KCLUST) {
        float acc = 0.f;
#pragma unroll 8
        for (int d = 0; d < DIM; ++d) { float df = zz[d] - cent[t * DIM + d]; acc += df * df; }
        d2[t] = acc;
    }
    __syncthreads();
    if (t == 0) {
        int best = 0; float bv = d2[0];
        for (int c = 1; c < KCLUST; ++c) if (d2[c] < bv) { bv = d2[c]; best = c; }
        ids[row] = best;
    }
}

// ---------------- two-level radix top-k select ----------------
#define TK_BINS 2048
__global__ void topk_select_kernel(const float* __restrict__ sims,
                                   const int* __restrict__ qcid,
                                   const int* __restrict__ qids,
                                   int* __restrict__ hard_idx,
                                   int* __restrict__ false_idx) {
    const int mode = blockIdx.x;
    const int row  = blockIdx.y;
    const int k    = (mode == 0) ? N_HARD : N_FALSE;
    int* out = (mode == 0) ? (hard_idx + (size_t)row * N_HARD)
                           : (false_idx + (size_t)row * N_FALSE);
    const float* s = sims + (size_t)row * QUEUE_N;
    const int myid = qids[row];

    __shared__ unsigned hist[TK_BINS];
    __shared__ unsigned sv[8];

    auto keyOf = [&](int j) -> unsigned {
        bool same = (qcid[j] == myid);
        bool keep = (mode == 0) ? !same : same;
        float v = keep ? s[j] : -1e9f;
        unsigned u = __float_as_uint(v);
        return (u & 0x80000000u) ? ~u : (u | 0x80000000u);
    };

    for (int b = threadIdx.x; b < TK_BINS; b += blockDim.x) hist[b] = 0;
    __syncthreads();
    for (int j = threadIdx.x; j < QUEUE_N; j += blockDim.x)
        atomicAdd(&hist[keyOf(j) >> 21], 1u);
    __syncthreads();
    if (threadIdx.x == 0) {
        unsigned cum = 0; int bsel = 0;
        for (int b = TK_BINS - 1; b >= 0; --b) { cum += hist[b]; if (cum >= (unsigned)k) { bsel = b; break; } }
        sv[0] = (unsigned)bsel; sv[1] = cum - hist[bsel];
    }
    __syncthreads();
    const unsigned bin1 = sv[0], above1 = sv[1];

    for (int b = threadIdx.x; b < TK_BINS; b += blockDim.x) hist[b] = 0;
    __syncthreads();
    for (int j = threadIdx.x; j < QUEUE_N; j += blockDim.x) {
        unsigned kk = keyOf(j);
        if ((kk >> 21) == bin1) atomicAdd(&hist[(kk >> 10) & 0x7FFu], 1u);
    }
    __syncthreads();
    if (threadIdx.x == 0) {
        unsigned need = (unsigned)k - above1;
        unsigned cum = 0; int bsel = 0;
        for (int b = TK_BINS - 1; b >= 0; --b) { cum += hist[b]; if (cum >= need) { bsel = b; break; } }
        sv[2] = (unsigned)bsel; sv[3] = cum - hist[bsel];
        sv[4] = 0; sv[5] = 0; sv[6] = 0;
    }
    __syncthreads();
    const unsigned bin2 = sv[2], above2 = sv[3];

    for (int j = threadIdx.x; j < QUEUE_N; j += blockDim.x) {
        unsigned kk  = keyOf(j);
        unsigned b1v = kk >> 21;
        if (b1v > bin1) {
            unsigned p = atomicAdd(&sv[4], 1u);
            out[p] = j;
        } else if (b1v == bin1) {
            unsigned b2v = (kk >> 10) & 0x7FFu;
            if (b2v > bin2) {
                unsigned p = atomicAdd(&sv[5], 1u);
                out[above1 + p] = j;
            } else if (b2v == bin2) {
                unsigned p = atomicAdd(&sv[6], 1u);
                unsigned slot = above1 + above2 + p;
                if (slot < (unsigned)k) out[slot] = j;
            }
        }
    }
}

// ---------------- gather: out[b, :] = queue[idx[b], :] ----------------
__global__ void gather_rows_kernel(const int* __restrict__ idx, const float* __restrict__ queue,
                                   float* __restrict__ out) {
    const int b = blockIdx.x, t = threadIdx.x;
    out[(size_t)b * DIM + t] = queue[(size_t)idx[b] * DIM + t];
}

// =============================================================================
extern "C" void kernel_launch(void* const* d_in, const int* in_sizes, int n_in,
                              void* d_out, int out_size, void* d_ws, size_t ws_size,
                              hipStream_t stream) {
    const float* f_q  = (const float*)d_in[0];
    const float* f_k1 = (const float*)d_in[1];
    const float* f_k2 = (const float*)d_in[2];
    const float* Wq1  = (const float*)d_in[3];
    const float* bq1  = (const float*)d_in[4];
    const float* Wq2  = (const float*)d_in[5];
    const float* bq2  = (const float*)d_in[6];
    const float* Wk1  = (const float*)d_in[7];
    const float* bk1  = (const float*)d_in[8];
    const float* Wk2  = (const float*)d_in[9];
    const float* bk2  = (const float*)d_in[10];
    const float* queue     = (const float*)d_in[11];
    const float* centroids = (const float*)d_in[12];
    const int*   qcid      = (const int*)d_in[13];

    float* out = (float*)d_out;
    char*  ws  = (char*)d_ws;

    bf16_t* Xq    = (bf16_t*)(ws + WS_XQ);
    bf16_t* Xk1   = (bf16_t*)(ws + WS_XK1);
    bf16_t* Xk2   = (bf16_t*)(ws + WS_XK2);
    bf16_t* W1qT  = (bf16_t*)(ws + WS_W1Q);
    bf16_t* W1kT  = (bf16_t*)(ws + WS_W1K);
    bf16_t* W2qT  = (bf16_t*)(ws + WS_W2Q);
    bf16_t* W2kT  = (bf16_t*)(ws + WS_W2K);
    float*  C1    = (float*)(ws + WS_C1);
    bf16_t* H     = (bf16_t*)(ws + WS_H);
    float*  C2    = (float*)(ws + WS_C2);
    bf16_t* Qbf   = (bf16_t*)(ws + WS_QBF);
    bf16_t* QueBf = (bf16_t*)(ws + WS_QUEBF);
    float*  Sims  = (float*)(ws + WS_SIMS);

    float* zq  = out + O_Q;
    float* zk1 = out + O_K1;
    float* zk2 = out + O_K2;
    int* qids   = (int*)(out + O_QIDS);
    int* k1ids  = (int*)(out + O_K1IDS);
    int* k2ids  = (int*)(out + O_K2IDS);
    int* h_idx  = (int*)(out + O_HIDX);
    int* f_idx  = (int*)(out + O_FIDX);

    // ---- conversions ----
    cvt_bf16_kernel<<<512, 256, 0, stream>>>(f_q,  Xq,  BATCH * HIDDEN);
    cvt_bf16_kernel<<<512, 256, 0, stream>>>(f_k1, Xk1, BATCH * HIDDEN);
    cvt_bf16_kernel<<<512, 256, 0, stream>>>(f_k2, Xk2, BATCH * HIDDEN);
    cvt_bf16_T_kernel<<<4096, 256, 0, stream>>>(Wq1, W1qT, HIDDEN, HIDDEN);
    cvt_bf16_T_kernel<<<4096, 256, 0, stream>>>(Wk1, W1kT, HIDDEN, HIDDEN);
    cvt_bf16_T_kernel<<<512,  256, 0, stream>>>(Wq2, W2qT, HIDDEN, DIM);
    cvt_bf16_T_kernel<<<512,  256, 0, stream>>>(Wk2, W2kT, HIDDEN, DIM);
    cvt_bf16_kernel<<<4096, 256, 0, stream>>>(queue, QueBf, QUEUE_N * DIM);

    // ---- 3 MLP branches ----
    struct Branch { const bf16_t* X; const bf16_t* W1; const float* b1;
                    const bf16_t* W2; const float* b2; float* z; bf16_t* zbf; int* ids; };
    Branch br[3] = {
        { Xq,  W1qT, bq1, W2qT, bq2, zq,  Qbf,     qids  },
        { Xk1, W1kT, bk1, W2kT, bk2, zk1, nullptr, k1ids },
        { Xk2, W1kT, bk1, W2kT, bk2, zk2, nullptr, k2ids },
    };
    for (int i = 0; i < 3; ++i) {
        // layer 1: [256,2048] = [256,2048] x [2048,2048]
        wmma_gemm_bf16_kernel<HIDDEN, 4><<<dim3(HIDDEN / 256, BATCH / 16), 128, 0, stream>>>(
            br[i].X, br[i].W1, C1, HIDDEN);
        bias_relu_bf16_kernel<<<1024, 256, 0, stream>>>(C1, br[i].b1, H, BATCH * HIDDEN, HIDDEN);
        // layer 2: [256,128] = [256,2048] x [2048,128]
        wmma_gemm_bf16_kernel<HIDDEN, 2><<<dim3(DIM / 128, BATCH / 16), 128, 0, stream>>>(
            H, br[i].W2, C2, DIM);
        bias_l2norm_kernel<<<BATCH, DIM, 0, stream>>>(C2, br[i].b2, br[i].z, br[i].zbf);
        assign_centroids_kernel<<<BATCH, DIM, 0, stream>>>(br[i].z, centroids, br[i].ids);
    }

    // ---- sims = q @ queue^T (queue rows are already B columns) ----
    wmma_gemm_bf16_kernel<DIM, 4><<<dim3(QUEUE_N / 256, BATCH / 16), 128, 0, stream>>>(
        Qbf, QueBf, Sims, QUEUE_N);

    // ---- cluster-masked top-k selection ----
    topk_select_kernel<<<dim3(2, BATCH), 256, 0, stream>>>(Sims, qcid, qids, h_idx, f_idx);

    // ---- gathers ----
    gather_rows_kernel<<<BATCH * N_HARD,  DIM, 0, stream>>>(h_idx, queue, out + O_HEMB);
    gather_rows_kernel<<<BATCH * N_FALSE, DIM, 0, stream>>>(f_idx, queue, out + O_FEMB);

    // ---- queue echo output ----
    hipMemcpyAsync(out + O_QUEUE, queue, (size_t)QUEUE_N * DIM * sizeof(float),
                   hipMemcpyDeviceToDevice, stream);
}
